// AttnBlock_67233418052323
// MI455X (gfx1250) — matrix operand order repaired and checked by reading, not compiled, not run on hardware
//
#include <hip/hip_runtime.h>
#include <hip/hip_bf16.h>
#include <math.h>

// ===========================================================================
// AttnBlock for MI455X (gfx1250, wave32, WMMA).
// Pipeline: GN stats -> GN apply + f16 pack -> w f16 pack -> qkv WMMA GEMMs
//           -> flash attention (WMMA S and PV, online softmax) -> proj WMMA
//           GEMM + bias + residual.
// All matrix math on v_wmma_f32_16x16x32_f16 (f32 accumulate).
// ===========================================================================

#define DEVINL __device__ __forceinline__

typedef __attribute__((ext_vector_type(16))) _Float16 v16h;
typedef __attribute__((ext_vector_type(8)))  _Float16 v8h;
typedef __attribute__((ext_vector_type(8)))  float    v8f;

static constexpr int Bn = 2;     // batch
static constexpr int C  = 512;   // channels
static constexpr int N  = 4096;  // spatial H*W
static constexpr int G  = 32;    // groups
static constexpr int CPG = C / G;

union V16U { v16h v; v8h h[2]; };

// A operand (16x32, f16), ISA 7.12.2 layout.
// lanes 0-15 : row m=lane,    K halves {0..7} then {16..23}
// lanes 16-31: row m=lane-16, K halves {8..15} then {24..31}
DEVINL v16h load_a16(const _Float16* p, int stride, int lane) {
  int m  = lane & 15;
  int k0 = (lane & 16) ? 8 : 0;
  const _Float16* q = p + (size_t)m * stride + k0;
  V16U u;
  u.h[0] = *(const v8h*)(q);
  u.h[1] = *(const v8h*)(q + 16);
  return u.v;
}

// B operand (32x16, f16) gathered from [n][k] (k-contiguous) storage.
// lanes 0-15: col n=lane, K {0..7} then {16..23} ; lanes 16-31: col n=lane-16, K {8..15} then {24..31}
DEVINL v16h load_b16(const _Float16* p, int stride, int lane) {
  int n  = lane & 15;
  int k0 = (lane & 16) ? 8 : 0;
  const _Float16* q = p + (size_t)n * stride + k0;
  V16U u;
  u.h[0] = *(const v8h*)(q);
  u.h[1] = *(const v8h*)(q + 16);
  return u.v;
}

DEVINL v8f wmma_f16(v16h a, v16h b, v8f c) {
  return __builtin_amdgcn_wmma_f32_16x16x32_f16(false, a, false, b,
                                                (short)0, c, false, false);
}

// ---------------------------------------------------------------------------
// 1) GroupNorm statistics: one block per (b, g); reduce 16*4096 elements.
// ---------------------------------------------------------------------------
__global__ void gn_stats_kernel(const float* __restrict__ x,
                                float* __restrict__ meanp,
                                float* __restrict__ rstdp) {
  int bg = blockIdx.x;              // 0..63
  int b = bg >> 5, g = bg & 31;
  const float* p = x + (size_t)b * C * N + (size_t)g * CPG * N;
  float s = 0.f, s2 = 0.f;
  for (int i = threadIdx.x; i < CPG * N; i += blockDim.x) {
    float v = p[i];
    s += v; s2 += v * v;
  }
  __shared__ float sh1[256], sh2[256];
  int tid = threadIdx.x;
  sh1[tid] = s; sh2[tid] = s2;
  __syncthreads();
  for (int st = 128; st > 0; st >>= 1) {
    if (tid < st) { sh1[tid] += sh1[tid + st]; sh2[tid] += sh2[tid + st]; }
    __syncthreads();
  }
  if (tid == 0) {
    const float inv = 1.f / (float)(CPG * N);
    float mu  = sh1[0] * inv;
    float var = sh2[0] * inv - mu * mu;
    meanp[bg] = mu;
    rstdp[bg] = rsqrtf(var + 1e-6f);
  }
}

// ---------------------------------------------------------------------------
// 2) Apply GN and pack f16, transposed to xnT[b][n][c] (c contiguous).
// ---------------------------------------------------------------------------
__global__ void gn_apply_kernel(const float* __restrict__ x,
                                const float* __restrict__ meanp,
                                const float* __restrict__ rstdp,
                                const float* __restrict__ gns,
                                const float* __restrict__ gnb,
                                _Float16* __restrict__ xnT) {
  size_t idx = (size_t)blockIdx.x * blockDim.x + threadIdx.x; // b*N*C + n*C + c
  int c = (int)(idx & (C - 1));
  size_t t = idx >> 9;
  int n = (int)(t & (N - 1));
  int b = (int)(t >> 12);
  int g = c >> 4;
  float v = x[(size_t)b * C * N + (size_t)c * N + n];
  float mu = meanp[b * G + g], rs = rstdp[b * G + g];
  v = (v - mu) * rs * gns[c] + gnb[c];
  xnT[idx] = (_Float16)v;
}

// ---------------------------------------------------------------------------
// 3) f32 -> f16 weight pack.
// ---------------------------------------------------------------------------
__global__ void f32_to_f16_kernel(const float* __restrict__ w,
                                  _Float16* __restrict__ o, int n) {
  int i = blockIdx.x * blockDim.x + threadIdx.x;
  if (i < n) o[i] = (_Float16)w[i];
}

// ---------------------------------------------------------------------------
// 4) qkv GEMM: out[o,n] = sum_c W[o,c] * xnT[n,c] + bias[o].
//    mode 0: store [n][o] f16 (for q/k).  mode 1: store [o][n] f16 (for v).
//    One wave per 16x16 tile; 8 waves per block; K = 512 = 16 chunks of 32.
// ---------------------------------------------------------------------------
__global__ void qkv_gemm_kernel(const _Float16* __restrict__ W,
                                const float* __restrict__ bias,
                                const _Float16* __restrict__ xnT,
                                _Float16* __restrict__ out, int mode) {
  int lane = threadIdx.x & 31;
  int warp = threadIdx.x >> 5;
  int tile = blockIdx.x * 8 + warp;        // 16384 tiles = B * 32 * 256
  int b   = tile >> 13;
  int rem = tile & 8191;
  int o0 = (rem >> 8) * 16;
  int n0 = (rem & 255) * 16;
  const _Float16* Xb = xnT + (size_t)b * N * C;

  v8f acc = {};
#pragma unroll
  for (int kk = 0; kk < 16; ++kk) {
    v16h a  = load_a16(W  + (size_t)o0 * C + kk * 32, C, lane);
    v16h bb = load_b16(Xb + (size_t)n0 * C + kk * 32, C, lane);
    acc = wmma_f16(a, bb, acc);
  }
#pragma unroll
  for (int r = 0; r < 8; ++r) {
    int m = r + ((lane >> 4) << 3);
    int n = lane & 15;
    float vfin = acc[r] + bias[o0 + m];
    if (mode == 0)
      out[(size_t)b * N * C + (size_t)(n0 + n) * C + (o0 + m)] = (_Float16)vfin;
    else
      out[(size_t)b * C * N + (size_t)(o0 + m) * N + (n0 + n)] = (_Float16)vfin;
  }
}

// ---------------------------------------------------------------------------
// 5) Flash attention. One wave (block of 32) per 16-query tile.
//    S = Q K^T (K=512), online softmax, O += P V^T (K=4096 in chunks of 32).
//    Q operands resident (128 VGPRs); O accumulators 32 tiles (256 VGPRs).
//    P (16x32) staged through 1 KB LDS to reshape D-layout -> A-layout.
// ---------------------------------------------------------------------------
__global__ void attn_kernel(const _Float16* __restrict__ qT,
                            const _Float16* __restrict__ kT,
                            const _Float16* __restrict__ vC,
                            _Float16* __restrict__ o16) {
  __shared__ _Float16 p_lds[16 * 32];
  const int lane = threadIdx.x & 31;
  const int b  = blockIdx.x >> 8;
  const int i0 = (blockIdx.x & 255) * 16;
  const float scale = 0.044194173824159216f; // 512^-0.5

  const _Float16* qb_ = qT + (size_t)b * N * C;
  const _Float16* kb_ = kT + (size_t)b * N * C;
  const _Float16* vb_ = vC + (size_t)b * C * N;

  // Resident Q A-operands: 16 chunks of K=32.
  v16h qa[16];
#pragma unroll
  for (int kk = 0; kk < 16; ++kk)
    qa[kk] = load_a16(qb_ + (size_t)i0 * C + kk * 32, C, lane);

  v8f oacc[32];
#pragma unroll
  for (int t = 0; t < 32; ++t) oacc[t] = {};
  float rowM[8], rowS[8], corrA[8];
#pragma unroll
  for (int r = 0; r < 8; ++r) { rowM[r] = -1e30f; rowS[r] = 0.f; }

  for (int jc = 0; jc < N / 32; ++jc) {
    // Two 16x16 S tiles covering keys [jc*32, jc*32+32).
    v8f s0 = {}, s1 = {};
#pragma unroll
    for (int kk = 0; kk < 16; ++kk) {
      v16h k0 = load_b16(kb_ + (size_t)(jc * 32 + 0)  * C + kk * 32, C, lane);
      s0 = wmma_f16(qa[kk], k0, s0);
    }
#pragma unroll
    for (int kk = 0; kk < 16; ++kk) {
      v16h k1 = load_b16(kb_ + (size_t)(jc * 32 + 16) * C + kk * 32, C, lane);
      s1 = wmma_f16(qa[kk], k1, s1);
    }

    // Online softmax per row; D layout: vgpr r holds row m=r+8*(lane/16),
    // col n=lane&15 -> row reductions are 16-lane shfl_xor butterflies.
#pragma unroll
    for (int r = 0; r < 8; ++r) {
      float a0 = s0[r] * scale, a1 = s1[r] * scale;
      float tmax = fmaxf(a0, a1);
#pragma unroll
      for (int off = 1; off < 16; off <<= 1)
        tmax = fmaxf(tmax, __shfl_xor(tmax, off, 32));
      float nm   = fmaxf(rowM[r], tmax);
      float corr = __expf(rowM[r] - nm);
      float e0 = __expf(a0 - nm), e1 = __expf(a1 - nm);
      float psum = e0 + e1;
#pragma unroll
      for (int off = 1; off < 16; off <<= 1)
        psum += __shfl_xor(psum, off, 32);
      rowS[r] = rowS[r] * corr + psum;
      rowM[r] = nm;
      corrA[r] = corr;
      int m  = r + ((lane >> 4) << 3);
      int jl = lane & 15;
      p_lds[m * 32 + jl]      = (_Float16)e0;
      p_lds[m * 32 + 16 + jl] = (_Float16)e1;
    }
    __syncthreads();
    v16h pa = load_a16(p_lds, 32, lane);
    __syncthreads();

    // O[t] = O[t]*corr + P * V^T chunk, for 32 channel tiles.
#pragma unroll
    for (int t = 0; t < 32; ++t) {
      v16h vv = load_b16(vb_ + (size_t)(t * 16) * N + jc * 32, N, lane);
#pragma unroll
      for (int r = 0; r < 8; ++r) oacc[t][r] *= corrA[r];
      oacc[t] = wmma_f16(pa, vv, oacc[t]);
    }
  }

  // Normalize by row sums and store O16[b][i][c] (c contiguous, f16).
  float rinv[8];
#pragma unroll
  for (int r = 0; r < 8; ++r) rinv[r] = 1.f / rowS[r];
  _Float16* ob = o16 + (size_t)b * N * C;
#pragma unroll
  for (int t = 0; t < 32; ++t) {
#pragma unroll
    for (int r = 0; r < 8; ++r) {
      int m  = r + ((lane >> 4) << 3);
      int cc = t * 16 + (lane & 15);
      ob[(size_t)(i0 + m) * C + cc] = (_Float16)(oacc[t][r] * rinv[r]);
    }
  }
}

// ---------------------------------------------------------------------------
// 6) proj GEMM + bias + residual: out[b,o,n] = x + pw @ O + pb (f32 out).
// ---------------------------------------------------------------------------
__global__ void proj_kernel(const _Float16* __restrict__ W,
                            const float* __restrict__ bias,
                            const _Float16* __restrict__ o16,
                            const float* __restrict__ x,
                            float* __restrict__ out) {
  int lane = threadIdx.x & 31;
  int warp = threadIdx.x >> 5;
  int tile = blockIdx.x * 8 + warp;
  int b   = tile >> 13;
  int rem = tile & 8191;
  int o0 = (rem >> 8) * 16;
  int n0 = (rem & 255) * 16;
  const _Float16* Ob = o16 + (size_t)b * N * C;

  v8f acc = {};
#pragma unroll
  for (int kk = 0; kk < 16; ++kk) {
    v16h a  = load_a16(W  + (size_t)o0 * C + kk * 32, C, lane);
    v16h bb = load_b16(Ob + (size_t)n0 * C + kk * 32, C, lane);
    acc = wmma_f16(a, bb, acc);
  }
#pragma unroll
  for (int r = 0; r < 8; ++r) {
    int m = r + ((lane >> 4) << 3);
    int n = lane & 15;
    size_t oi = (size_t)b * C * N + (size_t)(o0 + m) * N + (n0 + n);
    out[oi] = acc[r] + bias[o0 + m] + x[oi];
  }
}

// ---------------------------------------------------------------------------
// Host launcher. Workspace layout (bytes, 256-aligned):
//   mean[64], rstd[64], 4x w16 (512KB each), xnT/qT/kT/v/O16 (8MB each) ~44MB.
// ---------------------------------------------------------------------------
extern "C" void kernel_launch(void* const* d_in, const int* in_sizes, int n_in,
                              void* d_out, int out_size, void* d_ws, size_t ws_size,
                              hipStream_t stream) {
  (void)in_sizes; (void)n_in; (void)out_size; (void)ws_size;
  const float* x   = (const float*)d_in[0];
  const float* gns = (const float*)d_in[1];
  const float* gnb = (const float*)d_in[2];
  const float* qw  = (const float*)d_in[3];
  const float* qb  = (const float*)d_in[4];
  const float* kw  = (const float*)d_in[5];
  const float* kb  = (const float*)d_in[6];
  const float* vw  = (const float*)d_in[7];
  const float* vb  = (const float*)d_in[8];
  const float* pw  = (const float*)d_in[9];
  const float* pb  = (const float*)d_in[10];
  float* out = (float*)d_out;

  char* ws = (char*)d_ws;
  const size_t WSZ = (size_t)C * C * sizeof(_Float16);      // 512 KB
  const size_t TSZ = (size_t)Bn * N * C * sizeof(_Float16); // 8 MB
  float*    meanp = (float*)(ws + 0);
  float*    rstdp = (float*)(ws + 256);
  _Float16* w16q  = (_Float16*)(ws + 512);
  _Float16* w16k  = (_Float16*)(ws + 512 + WSZ);
  _Float16* w16v  = (_Float16*)(ws + 512 + 2 * WSZ);
  _Float16* w16p  = (_Float16*)(ws + 512 + 3 * WSZ);
  _Float16* xnT   = (_Float16*)(ws + 512 + 4 * WSZ);
  _Float16* qT    = (_Float16*)((char*)xnT + TSZ);
  _Float16* kT    = (_Float16*)((char*)xnT + 2 * TSZ);
  _Float16* vC    = (_Float16*)((char*)xnT + 3 * TSZ);
  _Float16* o16   = (_Float16*)((char*)xnT + 4 * TSZ);

  gn_stats_kernel<<<Bn * G, 256, 0, stream>>>(x, meanp, rstdp);
  gn_apply_kernel<<<(Bn * N * C) / 256, 256, 0, stream>>>(x, meanp, rstdp,
                                                          gns, gnb, xnT);
  f32_to_f16_kernel<<<(C * C + 255) / 256, 256, 0, stream>>>(qw, w16q, C * C);
  f32_to_f16_kernel<<<(C * C + 255) / 256, 256, 0, stream>>>(kw, w16k, C * C);
  f32_to_f16_kernel<<<(C * C + 255) / 256, 256, 0, stream>>>(vw, w16v, C * C);
  f32_to_f16_kernel<<<(C * C + 255) / 256, 256, 0, stream>>>(pw, w16p, C * C);

  // 16384 tiles each (B * 32 o-tiles * 256 n-tiles), 8 waves/block.
  qkv_gemm_kernel<<<2048, 256, 0, stream>>>(w16q, qb, xnT, qT, 0);
  qkv_gemm_kernel<<<2048, 256, 0, stream>>>(w16k, kb, xnT, kT, 0);
  qkv_gemm_kernel<<<2048, 256, 0, stream>>>(w16v, vb, xnT, vC, 1);

  attn_kernel<<<Bn * (N / 16), 32, 0, stream>>>(qT, kT, vC, o16);

  proj_kernel<<<2048, 256, 0, stream>>>(w16p, pb, o16, x, out);
}
